// MAT_31997506355868
// MI455X (gfx1250) — matrix operand order repaired and checked by reading, not compiled, not run on hardware
//
#include <hip/hip_runtime.h>
#include <hip/hip_bf16.h>

typedef __bf16 bf16;
typedef __attribute__((ext_vector_type(16))) __bf16 v16bf;
typedef __attribute__((ext_vector_type(8)))  __bf16 v8bf;
typedef __attribute__((ext_vector_type(8)))  float  v8f;

#define NT 1024
#define MROWS 4096          // B*S
#define NGROUP 64           // B*heads after flat reshape
#define GROUP_ELEMS 65536   // 1024 rows * 64 dph
#define ATT_SCALE 0.03125f  // 1/sqrt(1024)

// ---- WMMA fragment loaders (CDNA5 16-bit layouts, CDNA5 ISA) ----
// A (16x32): lane holds row M=lane%16; lanes 0-15 hold K {0..7,16..23},
// lanes 16-31 hold K {8..15,24..31}.
static __device__ __forceinline__ v16bf load_frag_a(const bf16* rowbase, int lane) {
  const int kk = (lane & 16) ? 8 : 0;
  union { v16bf v; v8bf h[2]; } u;
  u.h[0] = *(const v8bf*)(rowbase + kk);
  u.h[1] = *(const v8bf*)(rowbase + kk + 16);
  return u.v;
}
// B (32x16): lane holds column N=lane%16; lanes 0-15 hold K {0..7,16..23},
// lanes 16-31 hold K {8..15,24..31}.
static __device__ __forceinline__ v16bf load_frag_b(const bf16* colbase, int lane) {
  const int kk = (lane & 16) ? 8 : 0;
  union { v16bf v; v8bf h[2]; } u;
  u.h[0] = *(const v8bf*)(colbase + kk);
  u.h[1] = *(const v8bf*)(colbase + kk + 16);
  return u.v;
}
static __device__ __forceinline__ v8f wmma_bf16(v16bf a, v16bf b, v8f c) {
  return __builtin_amdgcn_wmma_f32_16x16x32_bf16(false, a, false, b, (short)0, c,
                                                 false, false);
}

// ---------------- small utility kernels ----------------
__global__ void k_zero(float* __restrict__ p, int n) {
  int i = blockIdx.x * blockDim.x + threadIdx.x;
  if (i < n) p[i] = 0.f;
}
__global__ void k_cvt(const float* __restrict__ src, bf16* __restrict__ dst, int n) {
  int i = blockIdx.x * blockDim.x + threadIdx.x;
  if (i < n) dst[i] = (bf16)src[i];
}

// ---------------- QKV projection: y = x @ W^T + b ----------------
// grid (32 rowblocks of 128, 16 colblocks of 64, 3 matrices), block 256 (8 waves)
// Software-pipelined: k-step fragments are loaded one iteration ahead so the
// WMMAs overlap the next step's global loads instead of stalling on loadcnt 0.
__global__ void k_qkv(const bf16* __restrict__ xb, const bf16* __restrict__ wb,
                      const float* __restrict__ bq, const float* __restrict__ bk,
                      const float* __restrict__ bv,
                      bf16* __restrict__ qb, bf16* __restrict__ kb,
                      bf16* __restrict__ vt) {
  const int lane = threadIdx.x & 31;
  const int wave = threadIdx.x >> 5;
  const int lm   = lane & 15;
  const int m0   = blockIdx.x * 128 + wave * 16;
  const int n0   = blockIdx.y * 64;
  const int mat  = blockIdx.z;
  const bf16*  W    = wb + (size_t)mat * NT * NT;
  const float* bias = (mat == 0) ? bq : (mat == 1) ? bk : bv;

  v8f acc[4] = {};
  const bf16* arow  = xb + (size_t)(m0 + lm) * NT;
  const bf16* brow0 = W  + (size_t)(n0 + lm) * NT;

  v16bf a_c = load_frag_a(arow, lane);
  v16bf b_c[4];
#pragma unroll
  for (int t = 0; t < 4; ++t)
    b_c[t] = load_frag_b(brow0 + (size_t)(t * 16) * NT, lane);

  for (int k0 = 0; k0 < NT; k0 += 32) {
    const int kn = (k0 + 32) & (NT - 1);   // wraps to 0 on last iter (discarded)
    __builtin_prefetch(arow + ((k0 + 64) & (NT - 1)), 0, 3);
    v16bf a_n = load_frag_a(arow + kn, lane);
    v16bf b_n[4];
#pragma unroll
    for (int t = 0; t < 4; ++t)
      b_n[t] = load_frag_b(brow0 + (size_t)(t * 16) * NT + kn, lane);
#pragma unroll
    for (int t = 0; t < 4; ++t)
      acc[t] = wmma_bf16(a_c, b_c[t], acc[t]);
    a_c = a_n;
#pragma unroll
    for (int t = 0; t < 4; ++t) b_c[t] = b_n[t];
  }

  const int roff = (lane & 16) ? 8 : 0;
#pragma unroll
  for (int t = 0; t < 4; ++t) {
    const int n = n0 + t * 16 + lm;
    const float bsn = bias[n];
#pragma unroll
    for (int jj = 0; jj < 8; ++jj) {
      const int m = m0 + roff + jj;
      const bf16 o = (bf16)(acc[t][jj] + bsn);
      if (mat == 0)      qb[(size_t)m * NT + n] = o;
      else if (mat == 1) kb[(size_t)m * NT + n] = o;
      else { // V stored transposed per group: vt[g][d][j]
        const int g = m >> 6;
        const int j = ((m & 63) << 4) + (n >> 6);
        const int d = n & 63;
        vt[(size_t)g * GROUP_ELEMS + d * 1024 + j] = o;
      }
    }
  }
}

// ---------------- Pass 1: softmax denominators Z_i ----------------
// grid 512 = 64 groups x 8 rowblocks(128); block 256 (8 waves x 16 rows)
__global__ void k_rowsum(const bf16* __restrict__ qb, const bf16* __restrict__ kb,
                         float* __restrict__ Z) {
  const int lane = threadIdx.x & 31;
  const int wave = threadIdx.x >> 5;
  const int lm   = lane & 15;
  const int g  = blockIdx.x >> 3;
  const int i0 = (blockIdx.x & 7) * 128 + wave * 16;
  const bf16* Qg = qb + (size_t)g * GROUP_ELEMS;
  const bf16* Kg = kb + (size_t)g * GROUP_ELEMS;
  const v16bf aq0 = load_frag_a(Qg + (i0 + lm) * 64, lane);
  const v16bf aq1 = load_frag_a(Qg + (i0 + lm) * 64 + 32, lane);
  float zacc[8] = {};

  v16bf b0_c = load_frag_b(Kg + lm * 64, lane);
  v16bf b1_c = load_frag_b(Kg + lm * 64 + 32, lane);
  for (int j0 = 0; j0 < 1024; j0 += 16) {
    const int jn = (j0 + 16) & 1023;
    const bf16* kn = Kg + (jn + lm) * 64;
    v16bf b0_n = load_frag_b(kn, lane);
    v16bf b1_n = load_frag_b(kn + 32, lane);
    v8f c = {};
    c = wmma_bf16(aq0, b0_c, c);
    c = wmma_bf16(aq1, b1_c, c);
#pragma unroll
    for (int jj = 0; jj < 8; ++jj)
      zacc[jj] += __expf(fmaxf(c[jj] * ATT_SCALE, 0.f));
    b0_c = b0_n; b1_c = b1_n;
  }
  const int roff = (lane & 16) ? 8 : 0;
#pragma unroll
  for (int jj = 0; jj < 8; ++jj) {
    float v = zacc[jj];
    v += __shfl_xor(v, 1);
    v += __shfl_xor(v, 2);
    v += __shfl_xor(v, 4);
    v += __shfl_xor(v, 8);
    if (lm == 0) Z[g * 1024 + i0 + roff + jj] = v;
  }
}

// ---------------- Pass 2: P = exp(relu(s/32))/Z, ctx = P@V, colsums ----------
__global__ void k_attn(const bf16* __restrict__ qb, const bf16* __restrict__ kb,
                       const bf16* __restrict__ vt, const float* __restrict__ Z,
                       float* __restrict__ ctx, float* __restrict__ at_sum) {
  __shared__ float colsum[1024];
  __shared__ __align__(16) bf16 pstage[8][512]; // per-wave 16x32 bf16 P tile
  const int tid  = threadIdx.x;
  const int lane = tid & 31;
  const int wave = tid >> 5;
  const int lm   = lane & 15;
  const int roff = (lane & 16) ? 8 : 0;
  const int g  = blockIdx.x >> 3;
  const int i0 = (blockIdx.x & 7) * 128 + wave * 16;

  for (int c = tid; c < 1024; c += 256) colsum[c] = 0.f;
  __syncthreads();

  const bf16* Qg = qb + (size_t)g * GROUP_ELEMS;
  const bf16* Kg = kb + (size_t)g * GROUP_ELEMS;
  const bf16* Vg = vt + (size_t)g * GROUP_ELEMS;
  const v16bf aq0 = load_frag_a(Qg + (i0 + lm) * 64, lane);
  const v16bf aq1 = load_frag_a(Qg + (i0 + lm) * 64 + 32, lane);
  float zinv[8];
  const float* zp = Z + g * 1024 + i0 + roff;
#pragma unroll
  for (int jj = 0; jj < 8; ++jj) zinv[jj] = 1.f / zp[jj];

  v8f acc[4] = {};
  bf16* ps = pstage[wave];

  // preload j0 = 0 fragments: K (2 subtiles x 2 chunks) and V (4 d-tiles)
  v16bf ka_c[2], kc_c[2], vf_c[4];
#pragma unroll
  for (int t = 0; t < 2; ++t) {
    const bf16* kc = Kg + (t * 16 + lm) * 64;
    ka_c[t] = load_frag_b(kc, lane);
    kc_c[t] = load_frag_b(kc + 32, lane);
  }
#pragma unroll
  for (int t = 0; t < 4; ++t)
    vf_c[t] = load_frag_b(Vg + (size_t)(t * 16 + lm) * 1024, lane);

  for (int j0 = 0; j0 < 1024; j0 += 32) {
    const int jn = (j0 + 32) & 1023;   // wraps on last iter (discarded)
    v16bf ka_n[2], kc_n[2], vf_n[4];
#pragma unroll
    for (int t = 0; t < 2; ++t) {
      const bf16* kc = Kg + (jn + t * 16 + lm) * 64;
      ka_n[t] = load_frag_b(kc, lane);
      kc_n[t] = load_frag_b(kc + 32, lane);
    }
#pragma unroll
    for (int t = 0; t < 4; ++t)
      vf_n[t] = load_frag_b(Vg + (size_t)(t * 16 + lm) * 1024 + jn, lane);

#pragma unroll
    for (int t = 0; t < 2; ++t) {
      const int jt = j0 + t * 16;
      v8f c = {};
      c = wmma_bf16(aq0, ka_c[t], c);
      c = wmma_bf16(aq1, kc_c[t], c);
      float s = 0.f;
#pragma unroll
      for (int jj = 0; jj < 8; ++jj) {
        float p = __expf(fmaxf(c[jj] * ATT_SCALE, 0.f)) * zinv[jj];
        s += p;
        ps[(roff + jj) * 32 + t * 16 + lm] = (bf16)p; // C-layout -> row-major LDS
      }
      s += __shfl_xor(s, 16); // combine the two row-halves of this column
      if (lane < 16) atomicAdd(&colsum[jt + lm], s);
    }
    // Same-wave LDS ops are in-order (DScnt); read P back in A-fragment layout.
    v16bf ap = load_frag_a(ps + lm * 32, lane);
#pragma unroll
    for (int t = 0; t < 4; ++t)
      acc[t] = wmma_bf16(ap, vf_c[t], acc[t]);

#pragma unroll
    for (int t = 0; t < 2; ++t) { ka_c[t] = ka_n[t]; kc_c[t] = kc_n[t]; }
#pragma unroll
    for (int t = 0; t < 4; ++t) vf_c[t] = vf_n[t];
  }

  float* crow = ctx + (size_t)g * GROUP_ELEMS;
#pragma unroll
  for (int t = 0; t < 4; ++t) {
#pragma unroll
    for (int jj = 0; jj < 8; ++jj)
      crow[(size_t)(i0 + roff + jj) * 64 + t * 16 + lm] = acc[t][jj];
  }
  __syncthreads();
  for (int c = tid; c < 1024; c += 256)
    atomicAdd(&at_sum[g * 1024 + c], colsum[c]);
}

// ---------------- residual + LayerNorm ----------------
__global__ void k_ln(const float* __restrict__ x, const float* __restrict__ ctx,
                     const float* __restrict__ gamma, const float* __restrict__ beta,
                     float* __restrict__ out) {
  __shared__ float red[16];
  const int tid  = threadIdx.x;
  const int lane = tid & 31;
  const int wave = tid >> 5;
  const size_t row = (size_t)blockIdx.x * NT;
  float h[4];
  float s = 0.f, s2 = 0.f;
#pragma unroll
  for (int q = 0; q < 4; ++q) {
    const int c = q * 256 + tid;
    const float v = x[row + c] + ctx[row + c];
    h[q] = v; s += v; s2 += v * v;
  }
#pragma unroll
  for (int off = 16; off >= 1; off >>= 1) {
    s  += __shfl_xor(s, off);
    s2 += __shfl_xor(s2, off);
  }
  if (lane == 0) { red[wave] = s; red[8 + wave] = s2; }
  __syncthreads();
  float S = 0.f, S2 = 0.f;
#pragma unroll
  for (int w = 0; w < 8; ++w) { S += red[w]; S2 += red[8 + w]; }
  const float mu   = S * (1.f / 1024.f);
  const float var  = S2 * (1.f / 1024.f) - mu * mu;
  const float rstd = rsqrtf(var + 1e-5f);
#pragma unroll
  for (int q = 0; q < 4; ++q) {
    const int c = q * 256 + tid;
    out[row + c] = (h[q] - mu) * rstd * gamma[c] + beta[c];
  }
}

extern "C" void kernel_launch(void* const* d_in, const int* in_sizes, int n_in,
                              void* d_out, int out_size, void* d_ws, size_t ws_size,
                              hipStream_t stream) {
  (void)in_sizes; (void)n_in; (void)out_size; (void)ws_size;
  const float* x     = (const float*)d_in[0];
  const float* Wq    = (const float*)d_in[1];
  const float* bq    = (const float*)d_in[2];
  const float* Wk    = (const float*)d_in[3];
  const float* bk    = (const float*)d_in[4];
  const float* Wv    = (const float*)d_in[5];
  const float* bv    = (const float*)d_in[6];
  const float* gamma = (const float*)d_in[7];
  const float* beta  = (const float*)d_in[8];
  float* out    = (float*)d_out;
  float* at_sum = out + (size_t)MROWS * NT; // (4,16,1024) flat after out

  char* ws = (char*)d_ws;
  bf16*  xb  = (bf16*)(ws);                              // 8 MB
  bf16*  wb  = (bf16*)(ws + (size_t)8  * 1024 * 1024);   // 6 MB (Wq|Wk|Wv)
  bf16*  qb  = (bf16*)(ws + (size_t)14 * 1024 * 1024);   // 8 MB
  bf16*  kb  = (bf16*)(ws + (size_t)22 * 1024 * 1024);   // 8 MB
  bf16*  vt  = (bf16*)(ws + (size_t)30 * 1024 * 1024);   // 8 MB (transposed)
  float* Z   = (float*)(ws + (size_t)38 * 1024 * 1024);  // 256 KB
  float* ctx = (float*)(ws + (size_t)39 * 1024 * 1024);  // 16 MB

  k_zero<<<256, 256, 0, stream>>>(at_sum, NGROUP * 1024);
  k_cvt<<<16384, 256, 0, stream>>>(x,  xb, MROWS * NT);
  k_cvt<<<4096,  256, 0, stream>>>(Wq, wb,               NT * NT);
  k_cvt<<<4096,  256, 0, stream>>>(Wk, wb + NT * NT,     NT * NT);
  k_cvt<<<4096,  256, 0, stream>>>(Wv, wb + 2 * NT * NT, NT * NT);

  dim3 gq(32, 16, 3);
  k_qkv<<<gq, 256, 0, stream>>>(xb, wb, bq, bk, bv, qb, kb, vt);
  k_rowsum<<<512, 256, 0, stream>>>(qb, kb, Z);
  k_attn<<<512, 256, 0, stream>>>(qb, kb, vt, Z, ctx, at_sum);
  k_ln<<<4096, 256, 0, stream>>>(x, ctx, gamma, beta, out);
}
